// get_model_23536420782432
// MI455X (gfx1250) — matrix operand order repaired and checked by reading, not compiled, not run on hardware
//
#include <hip/hip_runtime.h>
#include <hip/hip_bf16.h>
#include <math.h>

// ---------------------------------------------------------------------------
// PointNet++(MSG) + PCT part-segmentation forward for MI455X (gfx1250).
// All matmul-shaped work (1x1 convs, attention QK^T and V*attn^T) runs through
// one bf16 WMMA GEMM kernel (v_wmma_f32_16x16x32_bf16, f32 accumulate).
// Block tile 128x64 (4 waves x 32x64 wave tiles): 8 WMMAs per wave per K-step
// off 12 ds_load_b128 fragment loads (1.5 loads/WMMA).
// ---------------------------------------------------------------------------

static constexpr int B_   = 4;
static constexpr int N_   = 2048;
static constexpr int NCLS = 50;

typedef __attribute__((ext_vector_type(16))) __bf16 v16bf;
typedef __attribute__((ext_vector_type(8)))  float  v8f;

__device__ __forceinline__ __bf16 f2bf(float f) {
  unsigned u = __builtin_bit_cast(unsigned, f);
  u += 0x7FFFu + ((u >> 16) & 1u);              // round-to-nearest-even
  unsigned short h = (unsigned short)(u >> 16);
  return __builtin_bit_cast(__bf16, h);
}

// ---------------------------------------------------------------------------
// WMMA GEMM: C[M,N] = op(A)[M,K] * op(B)[K,N] (+bias[m]), batched on z.
// 128x64 block tile, K-step 32, 4 waves of 32 stacked on M.
// A/B staged f32->bf16 into LDS; B staged transposed [n][k] so fragment loads
// are contiguous 16B chunks matching the ISA 16-bit operand layouts.
// ---------------------------------------------------------------------------
__launch_bounds__(128)
__global__ void pn_gemm(const float* __restrict__ Ap, const float* __restrict__ Bp,
                        const float* __restrict__ bias, float* __restrict__ Cp,
                        int M, int N, int K, int lda, int ldb, int ldc,
                        int transA, int transB,
                        long long sA, long long sB, long long sC)
{
  __shared__ __align__(16) __bf16 sAt[128 * 40];  // row stride 40 bf16 = 80B
  __shared__ __align__(16) __bf16 sBt[64 * 40];   // stored [n][k]
  const float* Ab = Ap + (long long)blockIdx.z * sA;
  const float* Bb = Bp + (long long)blockIdx.z * sB;
  float*       Cb = Cp + (long long)blockIdx.z * sC;

  const int tid  = threadIdx.x;
  const int wave = tid >> 5;
  const int lane = tid & 31;
  const int hl   = lane >> 4;
  const int l15  = lane & 15;
  const int rowBase = blockIdx.y * 128;
  const int colBase = blockIdx.x * 64;
  const int wm = wave * 32;                       // wave row offset in block

  v8f acc[2][4] = {};

  for (int k0 = 0; k0 < K; k0 += 32) {
    // ---- stage A tile [128][32] (zero-padded at edges) ----
    for (int i = tid; i < 128 * 32; i += 128) {
      int m  = i >> 5;
      int kk = i & 31;
      int gm = rowBase + m, gk = k0 + kk;
      float v = 0.f;
      if (gm < M && gk < K)
        v = transA ? Ab[(long long)gk * lda + gm] : Ab[(long long)gm * lda + gk];
      sAt[m * 40 + kk] = f2bf(v);
    }
    // ---- stage B tile transposed into [n][k] ----
    if (!transB) {
      for (int i = tid; i < 64 * 32; i += 128) {
        int nn = i & 63;
        int kk = i >> 6;
        int gn = colBase + nn, gk = k0 + kk;
        float v = (gn < N && gk < K) ? Bb[(long long)gk * ldb + gn] : 0.f;
        sBt[nn * 40 + kk] = f2bf(v);
      }
    } else {
      for (int i = tid; i < 64 * 32; i += 128) {
        int kk = i & 31;
        int nn = i >> 5;
        int gn = colBase + nn, gk = k0 + kk;
        float v = (gn < N && gk < K) ? Bb[(long long)gn * ldb + gk] : 0.f;
        sBt[nn * 40 + kk] = f2bf(v);
      }
    }
    // speculative prefetch of the next K tile (global_prefetch_b8)
    if (k0 + 32 < K) {
      int pm = rowBase + tid;
      int pk = k0 + 32;
      if (pm < M && pk < K)
        __builtin_prefetch(transA ? &Ab[(long long)pk * lda + pm]
                                  : &Ab[(long long)pm * lda + pk], 0, 1);
      int pn = colBase + (tid >> 1);
      int pk2 = pk + ((tid & 1) << 4);
      if (pn < N && pk2 < K)
        __builtin_prefetch(transB ? &Bb[(long long)pn * ldb + pk2]
                                  : &Bb[(long long)pk2 * ldb + pn], 0, 1);
    }
    __syncthreads();

    union FA { v16bf v; uint4 q[2]; };
    FA fa[2], fb[4];
    // A fragment: lane holds row m, K chunks at {hl*8, 16+hl*8}  (16B each)
    #pragma unroll
    for (int mi = 0; mi < 2; ++mi) {
      const __bf16* r = &sAt[(wm + mi * 16 + l15) * 40];
      fa[mi].q[0] = *(const uint4*)(r + hl * 8);
      fa[mi].q[1] = *(const uint4*)(r + 16 + hl * 8);
    }
    // B fragment: lane holds column n, K chunks at {hl*8, 16+hl*8}  (16B each)
    #pragma unroll
    for (int ni = 0; ni < 4; ++ni) {
      const __bf16* c = &sBt[(ni * 16 + l15) * 40];
      fb[ni].q[0] = *(const uint4*)(c + hl * 8);
      fb[ni].q[1] = *(const uint4*)(c + 16 + hl * 8);
    }
    #pragma unroll
    for (int mi = 0; mi < 2; ++mi)
      #pragma unroll
      for (int ni = 0; ni < 4; ++ni)
        acc[mi][ni] = __builtin_amdgcn_wmma_f32_16x16x32_bf16(
            false, fa[mi].v, false, fb[ni].v, (short)0, acc[mi][ni], false, false);
    __syncthreads();
  }

  // C/D layout: VGPR j -> M = j + 8*hl, N = l15
  #pragma unroll
  for (int mi = 0; mi < 2; ++mi)
    #pragma unroll
    for (int ni = 0; ni < 4; ++ni)
      #pragma unroll
      for (int j = 0; j < 8; ++j) {
        int m = rowBase + wm + mi * 16 + j + 8 * hl;
        int n = colBase + ni * 16 + l15;
        if (m < M && n < N) {
          float v = acc[mi][ni][j];
          if (bias) v += bias[m];
          Cb[(long long)m * ldc + n] = v;
        }
      }
}

// ---------------------------------------------------------------------------
// BatchNorm (mean/var over batch*spatial per channel) + activation
// ---------------------------------------------------------------------------
__global__ void pn_bn_stats(const float* __restrict__ x, float* __restrict__ mean,
                            float* __restrict__ rstd, int nb, long long NN,
                            long long bstride)
{
  __shared__ float s1[256];
  __shared__ float s2[256];
  int c = blockIdx.x;
  float sum = 0.f, sq = 0.f;
  long long tot = (long long)nb * NN;
  for (long long i = threadIdx.x; i < tot; i += blockDim.x) {
    long long b = i / NN, n = i % NN;
    float v = x[b * bstride + (long long)c * NN + n];
    sum += v; sq += v * v;
  }
  s1[threadIdx.x] = sum; s2[threadIdx.x] = sq; __syncthreads();
  for (int s = 128; s > 0; s >>= 1) {
    if ((int)threadIdx.x < s) { s1[threadIdx.x] += s1[threadIdx.x + s]; s2[threadIdx.x] += s2[threadIdx.x + s]; }
    __syncthreads();
  }
  if (threadIdx.x == 0) {
    float m = s1[0] / (float)tot;
    float var = s2[0] / (float)tot - m * m;
    mean[c] = m;
    rstd[c] = rsqrtf(var + 1e-5f);
  }
}

__global__ void pn_bn_act(float* __restrict__ x, const float* __restrict__ mean,
                          const float* __restrict__ rstd, int C, long long NN,
                          long long tot, int act)
{
  long long i = (long long)blockIdx.x * blockDim.x + threadIdx.x;
  if (i >= tot) return;
  int c = (int)((i / NN) % C);
  float v = (x[i] - mean[c]) * rstd[c];
  if (act == 1)      v = fmaxf(v, 0.f);
  else if (act == 2) v = v > 0.f ? v : 0.2f * v;
  x[i] = v;
}

// ---------------------------------------------------------------------------
// Farthest point sampling: one block per batch, iterative argmax
// ---------------------------------------------------------------------------
__global__ void pn_fps(const float* __restrict__ xyz, int n, int S,
                       int* __restrict__ out, float* __restrict__ dist)
{
  int b = blockIdx.x;
  const float* X = xyz + (long long)b * 3 * n;
  float* D = dist + (long long)b * n;
  __shared__ float sv[256];
  __shared__ int   si[256];
  __shared__ int   sfar;
  int tid = threadIdx.x;
  for (int i = tid; i < n; i += 256) D[i] = 1e10f;
  if (tid == 0) sfar = 0;
  __syncthreads();
  for (int s = 0; s < S; ++s) {
    int far = sfar;
    if (tid == 0) out[b * S + s] = far;
    float cx = X[far], cy = X[n + far], cz = X[2 * n + far];
    float best = -1.f; int bi = 0;
    for (int i = tid; i < n; i += 256) {
      float dx = X[i] - cx, dy = X[n + i] - cy, dz = X[2 * n + i] - cz;
      float d = fminf(D[i], dx * dx + dy * dy + dz * dz);
      D[i] = d;
      if (d > best) { best = d; bi = i; }
    }
    sv[tid] = best; si[tid] = bi; __syncthreads();
    for (int t2 = 128; t2 > 0; t2 >>= 1) {
      if (tid < t2 && sv[tid + t2] > sv[tid]) { sv[tid] = sv[tid + t2]; si[tid] = si[tid + t2]; }
      __syncthreads();
    }
    if (tid == 0) sfar = si[0];
    __syncthreads();
  }
}

__global__ void pn_gather3(const float* __restrict__ xyz, const int* __restrict__ fidx,
                           int n, int S, float* __restrict__ nxyz, int tot)
{
  int t = blockIdx.x * blockDim.x + threadIdx.x;
  if (t >= tot) return;
  int b = t / S, s = t % S;
  int i = fidx[b * S + s];
  for (int j = 0; j < 3; ++j)
    nxyz[(long long)b * 3 * S + j * S + s] = xyz[(long long)b * 3 * n + (long long)j * n + i];
}

// Ball query: first K in-index-order points within radius (matches jnp.sort path)
__global__ void pn_ball(const float* __restrict__ xyz, const float* __restrict__ nxyz,
                        int n, int S, float r2, int K, int* __restrict__ gi, int tot)
{
  int t = blockIdx.x * blockDim.x + threadIdx.x;
  if (t >= tot) return;
  int b = t / S, s = t % S;
  const float* X = xyz + (long long)b * 3 * n;
  const float* C = nxyz + (long long)b * 3 * S;
  float cx = C[s], cy = C[S + s], cz = C[2 * S + s];
  int* out = gi + ((long long)b * S + s) * K;
  int cnt = 0, first = 0; bool have = false;
  for (int i = 0; i < n && cnt < K; ++i) {
    float dx = X[i] - cx, dy = X[n + i] - cy, dz = X[2 * n + i] - cz;
    if (dx * dx + dy * dy + dz * dz <= r2) {
      if (!have) { first = i; have = true; }
      out[cnt++] = i;
    }
  }
  for (; cnt < K; ++cnt) out[cnt] = first;
}

// Build grouped feature tensor [B][Cin+3][K*S]: point feats then relative xyz
__global__ void pn_group(const float* __restrict__ pts, const float* __restrict__ xyz,
                         const float* __restrict__ nxyz, const int* __restrict__ gi,
                         int Cin, int n, int S, int K, float* __restrict__ out, int tot)
{
  int t = blockIdx.x * blockDim.x + threadIdx.x;
  if (t >= tot) return;            // tot = B*S*K
  int b = t / (S * K);
  int r = t % (S * K);
  int s = r / K, k = r % K;
  int idx = gi[((long long)b * S + s) * K + k];
  long long NN = (long long)K * S;
  long long ob = (long long)b * (Cin + 3) * NN;
  long long col = (long long)k * S + s;
  for (int c = 0; c < Cin; ++c)
    out[ob + (long long)c * NN + col] = pts[(long long)b * Cin * n + (long long)c * n + idx];
  for (int j = 0; j < 3; ++j)
    out[ob + (long long)(Cin + j) * NN + col] =
        xyz[(long long)b * 3 * n + (long long)j * n + idx] - nxyz[(long long)b * 3 * S + j * S + s];
}

// Max over the K (sample) axis, writing into a channel-offset of dst
__global__ void pn_maxk(const float* __restrict__ in, float* __restrict__ out,
                        int C, int K, int S, int Cdst, int rowoff, long long tot)
{
  long long t = (long long)blockIdx.x * blockDim.x + threadIdx.x;
  if (t >= tot) return;            // tot = B*C*S
  long long b = t / ((long long)C * S);
  long long r = t % ((long long)C * S);
  long long c = r / S, s = r % S;
  const float* p = in + b * (long long)C * K * S + c * (long long)K * S + s;
  float m = -1e30f;
  for (int k = 0; k < K; ++k) m = fmaxf(m, p[(long long)k * S]);
  out[b * (long long)Cdst * S + (rowoff + c) * S + s] = m;
}

// ---------------------------------------------------------------------------
// Attention helpers
// ---------------------------------------------------------------------------
__global__ void pn_softmax_rows(float* __restrict__ a, int cols)
{
  long long row = blockIdx.x;
  float* p = a + row * (long long)cols;
  __shared__ float sb[128];
  int t = threadIdx.x;
  float mx = -1e30f;
  for (int i = t; i < cols; i += 128) mx = fmaxf(mx, p[i]);
  sb[t] = mx; __syncthreads();
  for (int s = 64; s > 0; s >>= 1) { if (t < s) sb[t] = fmaxf(sb[t], sb[t + s]); __syncthreads(); }
  mx = sb[0]; __syncthreads();
  float sum = 0.f;
  for (int i = t; i < cols; i += 128) { float e = __expf(p[i] - mx); p[i] = e; sum += e; }
  sb[t] = sum; __syncthreads();
  for (int s = 64; s > 0; s >>= 1) { if (t < s) sb[t] += sb[t + s]; __syncthreads(); }
  float inv = 1.f / sb[0];
  __syncthreads();
  for (int i = t; i < cols; i += 128) p[i] *= inv;
}

__global__ void pn_colnorm(float* __restrict__ a, int n, int tot)
{
  int t = blockIdx.x * blockDim.x + threadIdx.x;
  if (t >= tot) return;            // tot = B*n
  int b = t / n, m = t % n;
  float* p = a + (long long)b * n * n + m;
  float s = 0.f;
  for (int r = 0; r < n; ++r) s += p[(long long)r * n];
  float inv = 1.f / (1e-9f + s);
  for (int r = 0; r < n; ++r) p[(long long)r * n] *= inv;
}

__global__ void pn_sub(const float* __restrict__ a, const float* __restrict__ b,
                       float* __restrict__ c, long long tot)
{
  long long i = (long long)blockIdx.x * blockDim.x + threadIdx.x;
  if (i < tot) c[i] = a[i] - b[i];
}
__global__ void pn_addip(float* __restrict__ a, const float* __restrict__ b, long long tot)
{
  long long i = (long long)blockIdx.x * blockDim.x + threadIdx.x;
  if (i < tot) a[i] += b[i];
}

__global__ void pn_copy_rows(float* __restrict__ dst, const float* __restrict__ src,
                             int Cdst, int Csrc, long long n, int rowoff, long long tot)
{
  long long i = (long long)blockIdx.x * blockDim.x + threadIdx.x;
  if (i >= tot) return;            // tot = B*Csrc*n
  long long b = i / ((long long)Csrc * n);
  long long r = i % ((long long)Csrc * n);
  long long c = r / n, x = r % n;
  dst[b * (long long)Cdst * n + (rowoff + c) * n + x] = src[i];
}

__global__ void pn_slice_rows(float* __restrict__ dst, const float* __restrict__ src,
                              int Csrc, int Ck, int off, long long n, long long tot)
{
  long long i = (long long)blockIdx.x * blockDim.x + threadIdx.x;
  if (i >= tot) return;            // tot = B*Ck*n
  long long b = i / ((long long)Ck * n);
  long long r = i % ((long long)Ck * n);
  long long c = r / n, x = r % n;
  dst[i] = src[b * (long long)Csrc * n + (off + c) * n + x];
}

__global__ void pn_poolmaxavg(const float* __restrict__ x, float* __restrict__ xmax,
                              float* __restrict__ xavg, int C, int n, int tot)
{
  int t = blockIdx.x * blockDim.x + threadIdx.x;
  if (t >= tot) return;            // tot = B*C
  const float* p = x + (long long)t * n;
  float m = -1e30f, s = 0.f;
  for (int i = 0; i < n; ++i) { float v = p[i]; m = fmaxf(m, v); s += v; }
  xmax[t] = m; xavg[t] = s / (float)n;
}

// label branch: 16->64 matmul, BN over batch, leaky-relu 0.2
__global__ void pn_label(const float* __restrict__ w, const float* __restrict__ cl,
                         float* __restrict__ out, int nb)
{
  int o = threadIdx.x;             // 64 threads
  float v[8]; float s = 0.f;
  for (int b = 0; b < nb; ++b) {
    float a = 0.f;
    for (int c = 0; c < 16; ++c) a += w[o * 16 + c] * cl[b * 16 + c];
    v[b] = a; s += a;
  }
  float m = s / (float)nb, var = 0.f;
  for (int b = 0; b < nb; ++b) { float d = v[b] - m; var += d * d; }
  var /= (float)nb;
  float rs = rsqrtf(var + 1e-5f);
  for (int b = 0; b < nb; ++b) {
    float y = (v[b] - m) * rs;
    out[b * 64 + o] = y > 0.f ? y : 0.2f * y;
  }
}

// Build s1 input: [hf(1024) | bcast max(1024) | bcast avg(1024) | bcast cls(64)]
__global__ void pn_build_s1in(const float* __restrict__ hf, const float* __restrict__ xmax,
                              const float* __restrict__ xavg, const float* __restrict__ clsf,
                              long long n, float* __restrict__ out, long long tot)
{
  const int CT = 3136;
  long long i = (long long)blockIdx.x * blockDim.x + threadIdx.x;
  if (i >= tot) return;
  long long b = i / ((long long)CT * n);
  long long r = i % ((long long)CT * n);
  long long c = r / n, x = r % n;
  float v;
  if (c < 1024)      v = hf[b * 1024 * n + c * n + x];
  else if (c < 2048) v = xmax[b * 1024 + (c - 1024)];
  else if (c < 3072) v = xavg[b * 1024 + (c - 2048)];
  else               v = clsf[b * 64 + (c - 3072)];
  out[i] = v;
}

// 3-NN + inverse-distance weights
__global__ void pn_knn3(const float* __restrict__ x1, const float* __restrict__ x2,
                        int n1, int n2, int* __restrict__ idx3, float* __restrict__ w3, int tot)
{
  int t = blockIdx.x * blockDim.x + threadIdx.x;
  if (t >= tot) return;            // tot = B*n1
  int b = t / n1, i = t % n1;
  float px = x1[(long long)b * 3 * n1 + i];
  float py = x1[(long long)b * 3 * n1 + n1 + i];
  float pz = x1[(long long)b * 3 * n1 + 2 * n1 + i];
  float d0 = 1e30f, d1 = 1e30f, d2 = 1e30f;
  int i0 = 0, i1 = 0, i2 = 0;
  const float* X = x2 + (long long)b * 3 * n2;
  for (int j = 0; j < n2; ++j) {
    float dx = X[j] - px, dy = X[n2 + j] - py, dz = X[2 * n2 + j] - pz;
    float d = dx * dx + dy * dy + dz * dz;
    if (d < d0)      { d2 = d1; i2 = i1; d1 = d0; i1 = i0; d0 = d; i0 = j; }
    else if (d < d1) { d2 = d1; i2 = i1; d1 = d;  i1 = j; }
    else if (d < d2) { d2 = d;  i2 = j; }
  }
  float r0 = 1.f / (d0 + 1e-8f), r1 = 1.f / (d1 + 1e-8f), r2 = 1.f / (d2 + 1e-8f);
  float s = r0 + r1 + r2;
  long long base = ((long long)b * n1 + i) * 3;
  idx3[base] = i0; idx3[base + 1] = i1; idx3[base + 2] = i2;
  w3[base] = r0 / s; w3[base + 1] = r1 / s; w3[base + 2] = r2 / s;
}

__global__ void pn_interp(const float* __restrict__ p2, const int* __restrict__ idx3,
                          const float* __restrict__ w3, int C, int n1, int n2,
                          float* __restrict__ out, int Cdst, int rowoff, long long tot)
{
  long long t = (long long)blockIdx.x * blockDim.x + threadIdx.x;
  if (t >= tot) return;            // tot = B*C*n1
  long long b = t / ((long long)C * n1);
  long long r = t % ((long long)C * n1);
  long long c = r / n1, i = r % n1;
  long long base = (b * n1 + i) * 3;
  const float* p = p2 + b * (long long)C * n2 + c * n2;
  float v = p[idx3[base]] * w3[base] + p[idx3[base + 1]] * w3[base + 1]
          + p[idx3[base + 2]] * w3[base + 2];
  out[b * (long long)Cdst * n1 + (rowoff + c) * n1 + i] = v;
}

// log-softmax over channels + transpose to [B,N,C]
__global__ void pn_logsm(const float* __restrict__ lg, float* __restrict__ out,
                         int n, int tot)
{
  int t = blockIdx.x * blockDim.x + threadIdx.x;
  if (t >= tot) return;            // tot = B*n
  int b = t / n, i = t % n;
  const float* p = lg + (long long)b * NCLS * n + i;
  float mx = -1e30f;
  for (int c = 0; c < NCLS; ++c) mx = fmaxf(mx, p[(long long)c * n]);
  float s = 0.f;
  for (int c = 0; c < NCLS; ++c) s += __expf(p[(long long)c * n] - mx);
  float lse = mx + __logf(s);
  float* o = out + ((long long)b * n + i) * NCLS;
  for (int c = 0; c < NCLS; ++c) o[c] = p[(long long)c * n] - lse;
}

// ---------------------------------------------------------------------------
// Host-side orchestration
// ---------------------------------------------------------------------------
static hipStream_t g_st;
static float *g_mean = nullptr, *g_rstd = nullptr;

static inline int cdiv(long long a, int b) { return (int)((a + b - 1) / b); }

struct Ws {
  char* base; size_t cap; size_t off;
  float* f(long long elems) {
    size_t bytes = ((size_t)elems * 4 + 255) & ~(size_t)255;
    float* p = (float*)(base + off); off += bytes; return p;
  }
  int* i32(long long elems) {
    size_t bytes = ((size_t)elems * 4 + 255) & ~(size_t)255;
    int* p = (int*)(base + off); off += bytes; return p;
  }
  size_t mark() const { return off; }
  void reset(size_t m) { off = m; }
};

static void gemm(const float* A, const float* Bm, const float* bias, float* C,
                 int M, int N, int K, int lda, int ldb, int ldc, int tA, int tB,
                 long long sA, long long sB, long long sC)
{
  dim3 g(cdiv(N, 64), cdiv(M, 128), B_);
  pn_gemm<<<g, dim3(128), 0, g_st>>>(A, Bm, bias, C, M, N, K, lda, ldb, ldc, tA, tB, sA, sB, sC);
}

static void bn_act(float* x, int C, long long NN, int act)
{
  pn_bn_stats<<<C, 256, 0, g_st>>>(x, g_mean, g_rstd, B_, NN, (long long)C * NN);
  long long tot = (long long)B_ * C * NN;
  pn_bn_act<<<cdiv(tot, 256), 256, 0, g_st>>>(x, g_mean, g_rstd, C, NN, tot, act);
}

// weight conv (stride-0 A) + BN + act; bias omitted where it cancels under BN
static void conv_bn(const float* w, const float* x, float* y, int co, int ci, long long n, int act)
{
  gemm(w, x, nullptr, y, co, (int)n, ci, ci, (int)n, (int)n, 0, 0,
       0LL, (long long)ci * n, (long long)co * n);
  bn_act(y, co, n, act);
}

// ----- parameter parsing (top-level dict order; params in sorted-key order) --
struct PC { void* const* din; int i; const float* nx() { return (const float*)din[i++]; } };
struct ConvP { const float* w; const float* b; };
static ConvP convB(PC& pc) { ConvP c; c.b = pc.nx(); c.w = pc.nx(); return c; }
static ConvP convW(PC& pc) { ConvP c; c.b = nullptr; c.w = pc.nx(); return c; }
struct SaP { const float* qkw; const float* tb; const float* tw; const float* vb; const float* vw; };
static SaP readSa(PC& pc) {
  SaP s; s.qkw = pc.nx(); s.tb = pc.nx(); s.tw = pc.nx(); s.vb = pc.nx(); s.vw = pc.nx(); return s;
}
struct PctP { ConvP c1, c2, fuse, label, s1, s2; SaP sa[5]; };
static PctP readPct(PC& pc) {
  PctP p;
  p.c1 = convW(pc); p.c2 = convW(pc); p.fuse = convW(pc); p.label = convW(pc);
  p.s1 = convB(pc); p.s2 = convB(pc);
  for (int i = 0; i < 5; ++i) p.sa[i] = readSa(pc);
  return p;
}

// ----- PCT transformer block --------------------------------------------------
static void run_pct(Ws& ws, const PctP& p, const float* x, int cin, int cout, int n,
                    const float* cls_label, float* out)
{
  size_t mk = ws.mark();
  long long hn = (long long)128 * n;
  float* ha = ws.f((long long)B_ * hn);
  float* hb = ws.f((long long)B_ * hn);
  conv_bn(p.c1.w, x, ha, 128, cin, n, 1);
  conv_bn(p.c2.w, ha, hb, 128, 128, n, 1);
  float* h = hb;                                // current features [B,128,n]
  float* xs   = ws.f((long long)B_ * 640 * n);
  float* q    = ws.f((long long)B_ * hn);
  float* xv   = ws.f((long long)B_ * hn);
  float* xr   = ws.f((long long)B_ * hn);
  float* attn = ws.f((long long)B_ * n * n);
  long long htot = (long long)B_ * hn;

  for (int i = 0; i < 5; ++i) {
    const SaP& sa = p.sa[i];
    // q = Wqk h   (shared for query and key)
    gemm(sa.qkw, h, nullptr, q, 128, n, 128, 128, n, n, 0, 0, 0LL, hn, hn);
    // attn = q^T q
    gemm(q, q, nullptr, attn, n, n, 128, n, n, n, 1, 0, hn, hn, (long long)n * n);
    pn_softmax_rows<<<B_ * n, 128, 0, g_st>>>(attn, n);
    pn_colnorm<<<cdiv(B_ * n, 256), 256, 0, g_st>>>(attn, n, B_ * n);
    // xv = Wv h + bv   (bias survives: attn rows are not normalized to 1)
    gemm(sa.vw, h, sa.vb, xv, 128, n, 128, 128, n, n, 0, 0, 0LL, hn, hn);
    // xr = xv * attn^T
    gemm(xv, attn, nullptr, xr, 128, n, n, n, n, n, 0, 1, hn, (long long)n * n, hn);
    // t = Wt (h - xr); bn; relu; h += t   (t-bias cancels under BN)
    pn_sub<<<cdiv(htot, 256), 256, 0, g_st>>>(h, xr, q, htot);
    gemm(sa.tw, q, nullptr, xv, 128, n, 128, 128, n, n, 0, 0, 0LL, hn, hn);
    bn_act(xv, 128, n, 1);
    pn_addip<<<cdiv(htot, 256), 256, 0, g_st>>>(h, xv, htot);
    pn_copy_rows<<<cdiv(htot, 256), 256, 0, g_st>>>(xs, h, 640, 128, n, i * 128, htot);
  }

  float* hf = ws.f((long long)B_ * 1024 * n);
  conv_bn(p.fuse.w, xs, hf, 1024, 640, n, 2);   // leaky 0.2
  float* xmax = ws.f(B_ * 1024);
  float* xavg = ws.f(B_ * 1024);
  pn_poolmaxavg<<<cdiv(B_ * 1024, 256), 256, 0, g_st>>>(hf, xmax, xavg, 1024, n, B_ * 1024);
  float* clsf = ws.f(B_ * 64);
  pn_label<<<1, 64, 0, g_st>>>(p.label.w, cls_label, clsf, B_);
  float* s1in = ws.f((long long)B_ * 3136 * n);
  long long s1tot = (long long)B_ * 3136 * n;
  pn_build_s1in<<<cdiv(s1tot, 256), 256, 0, g_st>>>(hf, xmax, xavg, clsf, n, s1in, s1tot);
  float* s1o = ws.f((long long)B_ * 512 * n);
  conv_bn(p.s1.w, s1in, s1o, 512, 3136, n, 1);
  conv_bn(p.s2.w, s1o, out, cout, 512, n, 1);
  ws.reset(mk);
}

// ----- SA multi-scale grouping ------------------------------------------------
static void run_sa_msg(Ws& ws, const float* xyz, const float* pts, int n, int Cin,
                       int S, int nb, const float* radii, const int* Ks,
                       const ConvP (*mlps)[3], const int (*chans)[3],
                       float* nxyz, float* npts, int CoutTot)
{
  size_t mk = ws.mark();
  int*   fidx = ws.i32((long long)B_ * S);
  float* dist = ws.f((long long)B_ * n);
  pn_fps<<<B_, 256, 0, g_st>>>(xyz, n, S, fidx, dist);
  pn_gather3<<<cdiv(B_ * S, 256), 256, 0, g_st>>>(xyz, fidx, n, S, nxyz, B_ * S);

  int rowoff = 0;
  for (int br = 0; br < nb; ++br) {
    size_t mk2 = ws.mark();
    int K = Ks[br];
    float r2 = radii[br] * radii[br];
    long long NN = (long long)K * S;
    int* gi = ws.i32((long long)B_ * S * K);
    pn_ball<<<cdiv(B_ * S, 128), 128, 0, g_st>>>(xyz, nxyz, n, S, r2, K, gi, B_ * S);
    int Cg = Cin + 3;
    float* g = ws.f((long long)B_ * Cg * NN);
    pn_group<<<cdiv((long long)B_ * S * K, 256), 256, 0, g_st>>>(
        pts, xyz, nxyz, gi, Cin, n, S, K, g, B_ * S * K);
    int maxco = chans[br][0];
    if (chans[br][1] > maxco) maxco = chans[br][1];
    if (chans[br][2] > maxco) maxco = chans[br][2];
    float* pa = ws.f((long long)B_ * maxco * NN);
    float* pb = ws.f((long long)B_ * maxco * NN);
    const float* cur = g; int ci = Cg;
    for (int l = 0; l < 3; ++l) {
      int co = chans[br][l];
      float* o = (l & 1) ? pb : pa;
      conv_bn(mlps[br][l].w, cur, o, co, ci, NN, 1);  // mlp biases cancel under BN
      cur = o; ci = co;
    }
    long long mtot = (long long)B_ * ci * S;
    pn_maxk<<<cdiv(mtot, 256), 256, 0, g_st>>>(cur, npts, ci, K, S, CoutTot, rowoff, mtot);
    rowoff += ci;
    ws.reset(mk2);
  }
  ws.reset(mk);
}

// ----- feature propagation ----------------------------------------------------
static void run_fp(Ws& ws, const float* xyz1, int n1, const float* xyz2, int n2,
                   const float* pts1, int C1, const float* pts2, int C2,
                   const ConvP* mlp, const int* ch, float* out)
{
  size_t mk = ws.mark();
  int*   idx3 = ws.i32((long long)B_ * n1 * 3);
  float* w3   = ws.f((long long)B_ * n1 * 3);
  pn_knn3<<<cdiv(B_ * n1, 256), 256, 0, g_st>>>(xyz1, xyz2, n1, n2, idx3, w3, B_ * n1);
  int Cc = C1 + C2;
  float* cat = ws.f((long long)B_ * Cc * n1);
  long long t1 = (long long)B_ * C1 * n1;
  pn_copy_rows<<<cdiv(t1, 256), 256, 0, g_st>>>(cat, pts1, Cc, C1, n1, 0, t1);
  long long t2 = (long long)B_ * C2 * n1;
  pn_interp<<<cdiv(t2, 256), 256, 0, g_st>>>(pts2, idx3, w3, C2, n1, n2, cat, Cc, C1, t2);
  float* tmp = ws.f((long long)B_ * ch[0] * n1);
  const float* cur = cat; int ci = Cc;
  for (int l = 0; l < 2; ++l) {
    float* o = (l == 1) ? out : tmp;
    conv_bn(mlp[l].w, cur, o, ch[l], ci, n1, 1);      // fp biases cancel under BN
    cur = o; ci = ch[l];
  }
  ws.reset(mk);
}

// ---------------------------------------------------------------------------
extern "C" void kernel_launch(void* const* d_in, const int* in_sizes, int n_in,
                              void* d_out, int out_size, void* d_ws, size_t ws_size,
                              hipStream_t stream)
{
  (void)in_sizes; (void)n_in; (void)out_size;
  g_st = stream;
  const float* xyz = (const float*)d_in[0];   // [B,6,N]
  const float* cls = (const float*)d_in[1];   // [B,16]

  // ---- parse params (sorted-key flatten order) ----
  PC pc{d_in, 2};
  ConvP fp1m[2]; for (int i = 0; i < 2; ++i) fp1m[i] = convB(pc);
  ConvP fp2m[2]; for (int i = 0; i < 2; ++i) fp2m[i] = convB(pc);
  ConvP fp3m[2]; for (int i = 0; i < 2; ++i) fp3m[i] = convB(pc);
  ConvP head1 = convB(pc);
  ConvP head2 = convB(pc);
  PctP pct0 = readPct(pc);
  PctP pct1 = readPct(pc);
  PctP pct2 = readPct(pc);
  PctP pct3 = readPct(pc);
  ConvP sa1m[3][3]; for (int b = 0; b < 3; ++b) for (int l = 0; l < 3; ++l) sa1m[b][l] = convB(pc);
  ConvP sa2m[2][3]; for (int b = 0; b < 2; ++b) for (int l = 0; l < 3; ++l) sa2m[b][l] = convB(pc);
  ConvP sa3m[1][3]; for (int l = 0; l < 3; ++l) sa3m[0][l] = convB(pc);

  Ws ws{(char*)d_ws, ws_size, 0};
  g_mean = ws.f(4096);
  g_rstd = ws.f(4096);

  // ---- l0 ----
  float* l0xyz = ws.f((long long)B_ * 3 * N_);
  long long st = (long long)B_ * 3 * N_;
  pn_slice_rows<<<cdiv(st, 256), 256, 0, g_st>>>(l0xyz, xyz, 6, 3, 0, N_, st);

  // ---- SA encoders ----
  static const float rad1[3] = {0.1f, 0.2f, 0.4f};
  static const int   ks1[3]  = {32, 64, 128};
  static const int   ch1[3][3] = {{32, 64, 128}, {64, 64, 128}, {64, 96, 128}};
  float* l1xyz = ws.f((long long)B_ * 3 * 512);
  float* l1p   = ws.f((long long)B_ * 384 * 512);
  run_sa_msg(ws, l0xyz, xyz, N_, 6, 512, 3, rad1, ks1, sa1m, ch1, l1xyz, l1p, 384);

  static const float rad2[2] = {0.4f, 0.8f};
  static const int   ks2[2]  = {64, 128};
  static const int   ch2[2][3] = {{128, 128, 256}, {128, 196, 256}};
  float* l2xyz = ws.f((long long)B_ * 3 * 128);
  float* l2p   = ws.f((long long)B_ * 512 * 128);
  run_sa_msg(ws, l1xyz, l1p, 512, 384, 128, 2, rad2, ks2, sa2m, ch2, l2xyz, l2p, 512);

  static const float rad3[1] = {0.8f};
  static const int   ks3[1]  = {32};
  static const int   ch3[1][3] = {{256, 512, 1024}};
  float* l3xyz = ws.f((long long)B_ * 3 * 32);
  float* l3p   = ws.f((long long)B_ * 1024 * 32);
  run_sa_msg(ws, l2xyz, l2p, 128, 512, 32, 1, rad3, ks3, sa3m, ch3, l3xyz, l3p, 1024);

  // ---- PCT blocks + feature propagation (coarse -> fine) ----
  float* pct3o = ws.f((long long)B_ * 1024 * 32);
  run_pct(ws, pct3, l3p, 1024, 1024, 32, cls, pct3o);

  float* pct2o = ws.f((long long)B_ * 512 * 128);
  run_pct(ws, pct2, l2p, 512, 512, 128, cls, pct2o);
  static const int fch3[2] = {256, 256};
  float* fp3o = ws.f((long long)B_ * 256 * 128);
  run_fp(ws, l2xyz, 128, l3xyz, 32, pct2o, 512, pct3o, 1024, fp3m, fch3, fp3o);

  float* pct1o = ws.f((long long)B_ * 384 * 512);
  run_pct(ws, pct1, l1p, 384, 384, 512, cls, pct1o);
  static const int fch2[2] = {256, 128};
  float* fp2o = ws.f((long long)B_ * 128 * 512);
  run_fp(ws, l1xyz, 512, l2xyz, 128, pct1o, 384, fp3o, 256, fp2m, fch2, fp2o);

  float* pct0o = ws.f((long long)B_ * 128 * N_);
  run_pct(ws, pct0, xyz, 6, 128, N_, cls, pct0o);
  static const int fch1[2] = {128, 128};
  float* fp1o = ws.f((long long)B_ * 128 * N_);
  run_fp(ws, l0xyz, N_, l1xyz, 512, pct0o, 128, fp2o, 128, fp1m, fch1, fp1o);

  // ---- head ----
  float* feat = ws.f((long long)B_ * 128 * N_);
  conv_bn(head1.w, fp1o, feat, 128, 128, N_, 1);      // head1 bias cancels under BN
  float* logits = ws.f((long long)B_ * NCLS * N_);
  gemm(head2.w, feat, head2.b, logits, NCLS, N_, 128, 128, N_, N_, 0, 0,
       0LL, (long long)128 * N_, (long long)NCLS * N_);
  pn_logsm<<<cdiv(B_ * N_, 256), 256, 0, g_st>>>(logits, (float*)d_out, N_, B_ * N_);

  // second output: l3_points (pct3 output), concatenated flat after the logits
  hipMemcpyAsync((float*)d_out + (long long)B_ * N_ * NCLS, pct3o,
                 sizeof(float) * (size_t)B_ * 1024 * 32, hipMemcpyDeviceToDevice, stream);
}
